// VectorQuantizer_73555609911364
// MI455X (gfx1250) — matrix operand order repaired and checked by reading, not compiled, not run on hardware
//
#include <hip/hip_runtime.h>
#include <hip/hip_bf16.h>

typedef __attribute__((ext_vector_type(16))) _Float16 v16h;
typedef __attribute__((ext_vector_type(8)))  _Float16 v8h;
typedef __attribute__((ext_vector_type(2)))  _Float16 v2h;
typedef __attribute__((ext_vector_type(8)))  float    v8f;
typedef __attribute__((ext_vector_type(4)))  unsigned int u32x4;
typedef __attribute__((ext_vector_type(8)))  int      i32x8;
typedef __attribute__((ext_vector_type(4)))  int      i32x4;

#define NROWS 65536
#define DDIM  256
#define VDIM  1024
#define ROWS_PER_BLK 64
#define INV_TEMP 0.5f

// LDS: [0,128K) prob f16 ; [128K,192K) TDM buf0 ; [192K,256K) TDM buf1
#define LG_BYTES   (64 * VDIM * 2)
#define BUF_BYTES  (128 * DDIM * 2)
#define SMEM_BYTES (LG_BYTES + 2 * BUF_BYTES)

// d_ws layout: cbH f16 [1024][256] ; cbTH f16 [256][1024] ; c2 f32 [1024]
#define WS_CBH_OFF  0
#define WS_CBT_OFF  (VDIM * DDIM * 2)
#define WS_C2_OFF   (2 * VDIM * DDIM * 2)

__device__ __forceinline__ float ssq(float4 v) {
    return v.x * v.x + v.y * v.y + v.z * v.z + v.w * v.w;
}

// Issue a TDM 2D tile load: cbTH[0..255][c*128 .. +127] -> LDS at ldsOff.
// Tile rows (Y=d, 256) x 128 halfs (X), row stride 1024 halfs, data_size=2B.
__device__ __forceinline__ void tdm_load_chunk(const _Float16* cbTH, int c, unsigned ldsOff) {
    unsigned long long ga = (unsigned long long)(const void*)(cbTH + (size_t)c * 128);
    u32x4 g0;
    g0[0] = 1u;                                             // count=1, user mode
    g0[1] = ldsOff;                                         // lds_addr (bytes)
    g0[2] = (unsigned)ga;                                   // global_addr[31:0]
    g0[3] = (unsigned)((ga >> 32) & 0x1FFFFFFu) | (2u << 30); // global_addr[56:32] | type=2
    i32x8 g1;
    g1[0] = 0x00010000;     // workgroup_mask=0, data_size=1 (2 bytes)
    g1[1] = 0;              // tensor_dim0[15:0]=0 (dim0 = 0x100000: no OOB clip)
    g1[2] = (256 << 16) | 0x10; // tensor_dim0[31:16]=0x10, tensor_dim1=256
    g1[3] = (128 << 16);    // tensor_dim1 hi=0, tile_dim0=128
    g1[4] = 256;            // tile_dim1=256, tile_dim2=0
    g1[5] = 1024;           // tensor_dim0_stride = 1024 (halfs)
    g1[6] = 0;              // stride0 hi, tensor_dim1_stride lo
    g1[7] = 0;
    i32x4 z4 = {0, 0, 0, 0};
#if defined(__has_include)
#if __has_include(<hip/amd_detail/amd_gfx1250_TDM.h>)
    i32x8 z8 = {0, 0, 0, 0, 0, 0, 0, 0};
    __builtin_amdgcn_tensor_load_to_lds(g0, g1, z4, z4, z8, 0);   // clang-23 / therock: 6 args
#else
    __builtin_amdgcn_tensor_load_to_lds(g0, g1, z4, z4, 0);       // ROCm 7.2 clang-22: 5 args
#endif
#else
    __builtin_amdgcn_tensor_load_to_lds(g0, g1, z4, z4, 0);
#endif
}

// ---------------- one-shot codebook precompute: f16 row-major, f16 transposed, |c|^2 ----
__global__ __launch_bounds__(256)
void vq_precvt(const float* __restrict__ cb, _Float16* __restrict__ cbH,
               _Float16* __restrict__ cbTH, float* __restrict__ c2) {
    int v = blockIdx.x * 256 + threadIdx.x;        // 0..1023 codebook row
    const float4* src = (const float4*)(cb + (size_t)v * DDIM);
    float s = 0.f;
#pragma unroll 4
    for (int q = 0; q < DDIM / 4; ++q) {
        float4 f = src[q];
        s += ssq(f);
        int d = q * 4;
        _Float16 h0 = (_Float16)f.x, h1 = (_Float16)f.y;
        _Float16 h2 = (_Float16)f.z, h3 = (_Float16)f.w;
        cbH[(size_t)v * DDIM + d + 0] = h0;
        cbH[(size_t)v * DDIM + d + 1] = h1;
        cbH[(size_t)v * DDIM + d + 2] = h2;
        cbH[(size_t)v * DDIM + d + 3] = h3;
        cbTH[(size_t)(d + 0) * VDIM + v] = h0;     // transposed [d][v]
        cbTH[(size_t)(d + 1) * VDIM + v] = h1;
        cbTH[(size_t)(d + 2) * VDIM + v] = h2;
        cbTH[(size_t)(d + 3) * VDIM + v] = h3;
    }
    c2[v] = s;
}

// ---------------- fused VQ + gumbel softmax + reconstruction ---------------------------
__global__ __launch_bounds__(256, 1)
void vq_fused(const float* __restrict__ x,
              const _Float16* __restrict__ cbH,
              const _Float16* __restrict__ cbTH,
              const float* __restrict__ c2w,
              const float* __restrict__ gu,
              float* __restrict__ qout,
              float* __restrict__ pout)
{
    extern __shared__ char smem[];
    _Float16* lg  = (_Float16*)smem;                  // [64][1024] f16 logits -> prob
    _Float16* cbT = (_Float16*)(smem + LG_BYTES);     // 2 x [256][128] f16 TDM buffers

    const int tid    = threadIdx.x;
    const int lane   = tid & 31;
    const int wave   = tid >> 5;
    const int l16    = lane & 15;
    const int lhi    = lane >> 4;
    const int blkRow = blockIdx.x * ROWS_PER_BLK;

    // ======== Phase 1: d2 = |x|^2 + |c|^2 - 2 x.c  -> logits (f16, LDS) =======
    const int mtile = wave >> 1;
    const int vhalf = wave & 1;
    const int arow  = blkRow + mtile * 16 + l16;

    v16h a[8];
    float x2p = 0.f;
#pragma unroll
    for (int s = 0; s < 8; ++s) {
        const float* px = x + (size_t)arow * DDIM + 32 * s + 8 * lhi;
        float4 f0 = *(const float4*)(px);
        float4 f1 = *(const float4*)(px + 4);
        float4 f2 = *(const float4*)(px + 16);
        float4 f3 = *(const float4*)(px + 20);
        x2p += ssq(f0) + ssq(f1) + ssq(f2) + ssq(f3);
        a[s][0]  = (_Float16)f0.x; a[s][1]  = (_Float16)f0.y;
        a[s][2]  = (_Float16)f0.z; a[s][3]  = (_Float16)f0.w;
        a[s][4]  = (_Float16)f1.x; a[s][5]  = (_Float16)f1.y;
        a[s][6]  = (_Float16)f1.z; a[s][7]  = (_Float16)f1.w;
        a[s][8]  = (_Float16)f2.x; a[s][9]  = (_Float16)f2.y;
        a[s][10] = (_Float16)f2.z; a[s][11] = (_Float16)f2.w;
        a[s][12] = (_Float16)f3.x; a[s][13] = (_Float16)f3.y;
        a[s][14] = (_Float16)f3.z; a[s][15] = (_Float16)f3.w;
    }
    x2p += __shfl_xor(x2p, 16);
    float x2r[8];
#pragma unroll
    for (int i = 0; i < 8; ++i) x2r[i] = __shfl(x2p, i + 8 * lhi);

    for (int vt = vhalf * 32; vt < vhalf * 32 + 32; ++vt) {
        const int vcol = vt * 16 + l16;
        const _Float16* pc = cbH + (size_t)vcol * DDIM;
        __builtin_prefetch(pc + 16 * DDIM, 0, 0);     // next V tile
        float c2 = c2w[vcol];
        v8f acc = {};
#pragma unroll
        for (int s = 0; s < 8; ++s) {
            const _Float16* pb = pc + 32 * s + 8 * lhi;
            v8h b0 = *(const v8h*)(pb);
            v8h b1 = *(const v8h*)(pb + 16);
            v16h b;
#pragma unroll
            for (int j = 0; j < 8; ++j) { b[j] = b0[j]; b[j + 8] = b1[j]; }
            acc = __builtin_amdgcn_wmma_f32_16x16x32_f16(
                false, a[s], false, b, (short)0, acc, false, false);
        }
#pragma unroll
        for (int i = 0; i < 8; ++i) {
            int m   = i + 8 * lhi;
            int row = blkRow + mtile * 16 + m;
            float d2   = x2r[i] + c2 - 2.f * acc[i];
            float dist = sqrtf(fmaxf(d2, 1e-12f));
            float u = gu[(size_t)row * VDIM + vcol];
            u = fminf(fmaxf(u, 1e-10f), 1.f - 1e-10f);
            float g = -__logf(-__logf(u));
            lg[(size_t)(mtile * 16 + m) * VDIM + vcol] = (_Float16)((g - dist) * INV_TEMP);
        }
    }
    __syncthreads();

    // ======== Phase 1b: row softmax; f32 prob -> global, f16 prob in place ====
#pragma unroll
    for (int rr = 0; rr < 8; ++rr) {
        int r = wave * 8 + rr;
        v2h* lrow = (v2h*)(lg + (size_t)r * VDIM);
        float mx = -3.0e38f;
        for (int t = lane; t < VDIM / 2; t += 32) {
            v2h h = lrow[t];
            mx = fmaxf(mx, fmaxf((float)h[0], (float)h[1]));
        }
#pragma unroll
        for (int off = 16; off > 0; off >>= 1) mx = fmaxf(mx, __shfl_xor(mx, off));
        float sm = 0.f;
        for (int t = lane; t < VDIM / 2; t += 32) {
            v2h h = lrow[t];
            sm += __expf((float)h[0] - mx) + __expf((float)h[1] - mx);
        }
#pragma unroll
        for (int off = 16; off > 0; off >>= 1) sm += __shfl_xor(sm, off);
        float inv = 1.0f / sm;
        float* prow = pout + (size_t)(blkRow + r) * VDIM;
        for (int t = lane; t < VDIM / 2; t += 32) {
            v2h h = lrow[t];
            float p0 = __expf((float)h[0] - mx) * inv;
            float p1 = __expf((float)h[1] - mx) * inv;
            float2 pv; pv.x = p0; pv.y = p1;
            ((float2*)prow)[t] = pv;
            v2h o; o[0] = (_Float16)p0; o[1] = (_Float16)p1;
            lrow[t] = o;                       // in place: same elem, same thread
        }
    }

    // ======== Phase 2: quantized = prob(f16) @ codebook, TDM double-buffered ==
    const int mt2    = wave >> 1;
    const int ntBase = (wave & 1) * 8;
    v8f qacc[8];
#pragma unroll
    for (int i = 0; i < 8; ++i) qacc[i] = (v8f){};

    if (wave == 0) tdm_load_chunk(cbTH, 0, LG_BYTES);           // prime buf0
    for (int c = 0; c < 8; ++c) {
        if (wave == 0) __builtin_amdgcn_s_wait_tensorcnt(0);    // chunk c landed
        __syncthreads();
        if (wave == 0 && c < 7)                                 // overlap copy of c+1
            tdm_load_chunk(cbTH, c + 1, LG_BYTES + ((c + 1) & 1) * BUF_BYTES);

        const _Float16* cbuf = cbT + (size_t)(c & 1) * (128 * DDIM);
#pragma unroll
        for (int ks = 0; ks < 4; ++ks) {
            const _Float16* pa = lg + (size_t)(mt2 * 16 + l16) * VDIM + c * 128 + 32 * ks + 8 * lhi;
            v8h alo = *(const v8h*)(pa);
            v8h ahi = *(const v8h*)(pa + 16);
            v16h af;
#pragma unroll
            for (int j = 0; j < 8; ++j) { af[j] = alo[j]; af[j + 8] = ahi[j]; }
#pragma unroll
            for (int tI = 0; tI < 8; ++tI) {
                int d = (ntBase + tI) * 16 + l16;
                const _Float16* pb = cbuf + (size_t)d * 128 + 32 * ks + 8 * lhi;
                v8h blo = *(const v8h*)(pb);
                v8h bhi = *(const v8h*)(pb + 16);
                v16h bf;
#pragma unroll
                for (int j = 0; j < 8; ++j) { bf[j] = blo[j]; bf[j + 8] = bhi[j]; }
                qacc[tI] = __builtin_amdgcn_wmma_f32_16x16x32_f16(
                    false, af, false, bf, (short)0, qacc[tI], false, false);
            }
        }
    }
    if (wave == 0) __builtin_amdgcn_s_wait_tensorcnt(0);        // drain before exit

#pragma unroll
    for (int tI = 0; tI < 8; ++tI) {
        int dcol = (ntBase + tI) * 16 + l16;
#pragma unroll
        for (int i = 0; i < 8; ++i) {
            int row = blkRow + mt2 * 16 + i + 8 * lhi;
            qout[(size_t)row * DDIM + dcol] = qacc[tI][i];
        }
    }
}

extern "C" void kernel_launch(void* const* d_in, const int* in_sizes, int n_in,
                              void* d_out, int out_size, void* d_ws, size_t ws_size,
                              hipStream_t stream) {
    const float* x  = (const float*)d_in[0];   // [16,4096,256]
    const float* cb = (const float*)d_in[1];   // [1024,256]
    const float* gu = (const float*)d_in[2];   // [65536,1024]
    float* qout = (float*)d_out;
    float* pout = (float*)d_out + (size_t)NROWS * DDIM;

    char* ws = (char*)d_ws;                    // needs ~1.03 MB scratch
    _Float16* cbH  = (_Float16*)(ws + WS_CBH_OFF);
    _Float16* cbTH = (_Float16*)(ws + WS_CBT_OFF);
    float*    c2   = (float*)(ws + WS_C2_OFF);

    (void)hipFuncSetAttribute((const void*)vq_fused,
                              hipFuncAttributeMaxDynamicSharedMemorySize, SMEM_BYTES);

    vq_precvt<<<dim3(VDIM / 256), 256, 0, stream>>>(cb, cbH, cbTH, c2);
    vq_fused<<<dim3(NROWS / ROWS_PER_BLK), 256, SMEM_BYTES, stream>>>(
        x, cbH, cbTH, c2, gu, qout, pout);
}
